// edge_agg_layer_77197742178844
// MI455X (gfx1250) — matrix operand order repaired and checked by reading, not compiled, not run on hardware
//
#include <hip/hip_runtime.h>
#include <hip/hip_bf16.h>
#include <math.h>
#include <stdint.h>

typedef __bf16 bf16;
typedef __attribute__((ext_vector_type(16))) __bf16 v16bf;
typedef __attribute__((ext_vector_type(8)))  float  v8f;

#define DMODEL 128
#define ROWB   (DMODEL * 4u)   // bytes per feature row (512)

// uniform base + 32-bit byte offset -> global_load/store saddr form
static __device__ __forceinline__ const float* cptr(const void* base, uint32_t off) {
    return (const float*)((const char*)base + off);
}
static __device__ __forceinline__ float* mptr(void* base, uint32_t off) {
    return (float*)((char*)base + off);
}
static __device__ __forceinline__ float4 ld4o(const void* base, uint32_t off) {
    return *(const float4*)((const char*)base + off);
}

static __device__ __forceinline__ void pack8(v16bf& a, int base, float4 u, float4 v) {
    a[base + 0] = (bf16)u.x; a[base + 1] = (bf16)u.y;
    a[base + 2] = (bf16)u.z; a[base + 3] = (bf16)u.w;
    a[base + 4] = (bf16)v.x; a[base + 5] = (bf16)v.y;
    a[base + 6] = (bf16)v.z; a[base + 7] = (bf16)v.w;
}

// ---------------------------------------------------------------------------
// Kernel 1: fuse stacked bias-free linears into one 128x128 bf16 weight each.
// Wf1 = W_edge2 @ W_edge ; Wf2 = W_lin2 @ W_lin   (so  h = x @ Wf^T)
// ---------------------------------------------------------------------------
__global__ void fuse_weights(const float* __restrict__ We2, const float* __restrict__ We,
                             const float* __restrict__ Wl2, const float* __restrict__ Wl,
                             bf16* __restrict__ Wf1, bf16* __restrict__ Wf2) {
    int n = blockIdx.x;
    int k = threadIdx.x;
    const float* L; const float* R; bf16* O;
    if (n < DMODEL) { L = We2; R = We; O = Wf1; }
    else            { n -= DMODEL; L = Wl2; R = Wl; O = Wf2; }
    float acc = 0.f;
    #pragma unroll 8
    for (int j = 0; j < DMODEL; ++j)
        acc += L[n * DMODEL + j] * R[j * DMODEL + k];
    O[n * DMODEL + k] = (bf16)acc;
}

// ---------------------------------------------------------------------------
// Kernel 2: scatter-sum cycle features: S[cycle_dst[m]] += cycle[cycle_src[m]]
// One wave per membership; each lane owns 4 columns -> global_atomic_add_f32.
// ---------------------------------------------------------------------------
__global__ void scatter_cycles(const float* __restrict__ cycle,
                               const int* __restrict__ csrc,
                               const int* __restrict__ cdst,
                               float* __restrict__ S, int Mtot) {
    int wave = (int)((blockIdx.x * (unsigned)blockDim.x + threadIdx.x) >> 5);
    int lane = threadIdx.x & 31;
    if (wave >= Mtot) return;
    const uint32_t soff = (uint32_t)csrc[wave] * ROWB + (uint32_t)lane * 16u;
    const uint32_t doff = (uint32_t)cdst[wave] * ROWB + (uint32_t)lane * 16u;
    const float4 v = ld4o(cycle, soff);
    float* dst = mptr(S, doff);
    unsafeAtomicAdd(dst + 0, v.x);
    unsafeAtomicAdd(dst + 1, v.y);
    unsafeAtomicAdd(dst + 2, v.z);
    unsafeAtomicAdd(dst + 3, v.w);
}

// ---------------------------------------------------------------------------
// Core bf16 WMMA GEMM: each wave computes a 16x128 row-tile of  X @ W^T,
// K = 128 as 4 x v_wmma_f32_16x16x32_bf16, 8 column tiles.
// EPI==1: epilogue adds bond gathers + scattered S content, RMW into S.
// EPI==2: epilogue writes fp32 result to out, reduces per-column sum/sumsq
//         via LDS ds_add_f32, then one global atomic per column per block.
// All per-lane addresses are uniform-base + u32 byte offsets (buffers < 4GB).
// ---------------------------------------------------------------------------
template <int EPI>
__global__ __launch_bounds__(256) void
gemm_bf16(const float* __restrict__ Xf, const bf16* __restrict__ W,
          const float* __restrict__ bond, const int* __restrict__ eidx,
          long long E, float* __restrict__ S, float* __restrict__ outp,
          float* __restrict__ stats) {
    __shared__ float sh[2 * DMODEL];

    const int tid  = threadIdx.x;
    const int lane = tid & 31;
    const int wv   = tid >> 5;

    if constexpr (EPI == 2) { sh[tid] = 0.f; __syncthreads(); }

    const long long waveG  = (long long)blockIdx.x * 8 + wv;
    const long long nWaves = E >> 4;                 // E % 16 == 0
    const bool active      = waveG < nWaves;
    const uint32_t rowBase = (uint32_t)(waveG << 4); // < 2^20, safe in u32

    const int o    = (lane >> 4) << 4;   // A K-offset: 0 or 16
    const int hi   = (lane >> 4) << 4;   // B K-offset: 0 or 16
    const int mofs = (lane >> 4) << 3;   // C row-offset: 0 or 8
    const int nl   = lane & 15;

    if (active) {
        // ---- A fragments (fp32 -> bf16 in-register), 4 K-steps of 32 ----
        v16bf afrag[4];
        const uint32_t xoff = (rowBase + (uint32_t)nl) * ROWB;
        #pragma unroll
        for (int ks = 0; ks < 4; ++ks) {
            const uint32_t kb = xoff + (uint32_t)(ks * 32 + o) * 4u;
            float4 c0a = ld4o(Xf, kb);
            float4 c0b = ld4o(Xf, kb + 16u);
            float4 c1a = ld4o(Xf, kb + 32u);
            float4 c1b = ld4o(Xf, kb + 48u);
            v16bf a;
            pack8(a, 0, c0a, c0b);
            pack8(a, 8, c1a, c1b);
            afrag[ks] = a;
        }

        // ---- edge endpoint byte offsets (EPI 1 only), hoisted over N tiles ----
        uint32_t off0[8], off1[8];
        if constexpr (EPI == 1) {
            const uint32_t ebase = rowBase + (uint32_t)mofs;
            #pragma unroll
            for (int r = 0; r < 8; ++r) {
                off0[r] = (uint32_t)eidx[ebase + r] * ROWB;
                off1[r] = (uint32_t)eidx[(size_t)E + ebase + r] * ROWB;
            }
        }
        const uint32_t rbase = (rowBase + (uint32_t)mofs) * ROWB;

        #pragma unroll
        for (int nt = 0; nt < 8; ++nt) {
            const int n = nt * 16 + nl;
            const uint32_t n4 = (uint32_t)n * 4u;
            const bf16* wrow = W + n * DMODEL;
            v8f c = {0.f, 0.f, 0.f, 0.f, 0.f, 0.f, 0.f, 0.f};
            #pragma unroll
            for (int ks = 0; ks < 4; ++ks) {
                // 16 contiguous bf16 of the fused weight row = one B fragment
                v16bf b = *(const v16bf*)(wrow + ks * 32 + hi);
                c = __builtin_amdgcn_wmma_f32_16x16x32_bf16(
                        false, afrag[ks], false, b, (short)0, c, false, false);
            }

            if constexpr (EPI == 1) {
                #pragma unroll
                for (int r = 0; r < 8; ++r) {
                    const float g = *cptr(bond, off0[r] + n4) +
                                    *cptr(bond, off1[r] + n4);
                    float* sp = mptr(S, rbase + (uint32_t)r * ROWB + n4);
                    *sp = *sp + c[r] + g;     // S already holds cycle scatter
                }
            } else {
                float cs = 0.f, cq = 0.f;
                #pragma unroll
                for (int r = 0; r < 8; ++r) {
                    const float v = c[r];
                    *mptr(outp, rbase + (uint32_t)r * ROWB + n4) = v;
                    cs += v;
                    cq += v * v;
                }
                atomicAdd(&sh[n], cs);            // ds_add_f32
                atomicAdd(&sh[DMODEL + n], cq);
            }
        }
    }

    if constexpr (EPI == 2) {
        __syncthreads();
        if (tid < DMODEL) {
            unsafeAtomicAdd(&stats[tid], sh[tid]);
            unsafeAtomicAdd(&stats[DMODEL + tid], sh[DMODEL + tid]);
        }
    }
}

// ---------------------------------------------------------------------------
// Kernel 5: BatchNorm (batch stats, biased var) + exact erf GELU, in place.
// ---------------------------------------------------------------------------
__global__ void bn_gelu(float* __restrict__ out, const float* __restrict__ stats,
                        const float* __restrict__ gamma, const float* __restrict__ beta,
                        long long total, float invE) {
    const long long i = ((long long)blockIdx.x * blockDim.x + threadIdx.x) * 4;
    if (i >= total) return;
    const uint32_t off = (uint32_t)i * 4u;       // total*4 bytes = 512MB < 4GB
    const int n = (int)(i & (DMODEL - 1));
    float4 v = ld4o(out, off);
    float r[4] = {v.x, v.y, v.z, v.w};
    #pragma unroll
    for (int j = 0; j < 4; ++j) {
        const float mean = stats[n + j] * invE;
        const float var  = stats[DMODEL + n + j] * invE - mean * mean;
        float x = (r[j] - mean) * rsqrtf(var + 1e-5f) * gamma[n + j] + beta[n + j];
        r[j] = 0.5f * x * (1.0f + erff(x * 0.70710678118654752f));
    }
    v.x = r[0]; v.y = r[1]; v.z = r[2]; v.w = r[3];
    *(float4*)((char*)out + off) = v;
}

// ---------------------------------------------------------------------------
extern "C" void kernel_launch(void* const* d_in, const int* in_sizes, int n_in,
                              void* d_out, int out_size, void* d_ws, size_t ws_size,
                              hipStream_t stream) {
    const float* x       = (const float*)d_in[0];
    const float* bond    = (const float*)d_in[1];
    const float* cycle   = (const float*)d_in[2];
    const float* W_edge  = (const float*)d_in[3];
    const float* W_edge2 = (const float*)d_in[4];
    const float* W_lin   = (const float*)d_in[5];
    const float* W_lin2  = (const float*)d_in[6];
    const float* gamma   = (const float*)d_in[7];
    const float* beta    = (const float*)d_in[8];
    const int*   eidx    = (const int*)d_in[9];
    const int*   csrc    = (const int*)d_in[10];
    const int*   cdst    = (const int*)d_in[11];
    float* out = (float*)d_out;

    const long long E = in_sizes[0] / DMODEL;   // 1,000,000 (multiple of 16)
    const int Mtot    = in_sizes[10];           // 1,000,000 memberships

    char* ws = (char*)d_ws;
    bf16*  Wf1   = (bf16*)(ws);                 //  32 KB
    bf16*  Wf2   = (bf16*)(ws + 32768);         //  32 KB
    float* stats = (float*)(ws + 65536);        //   1 KB (sum | sumsq)
    float* S     = (float*)(ws + (1 << 20));    // E*128 fp32 accumulator

    hipMemsetAsync(S, 0, (size_t)E * DMODEL * sizeof(float), stream);
    hipMemsetAsync(stats, 0, 2 * DMODEL * sizeof(float), stream);

    fuse_weights<<<2 * DMODEL, DMODEL, 0, stream>>>(W_edge2, W_edge, W_lin2, W_lin, Wf1, Wf2);

    {
        const long long thr = (long long)Mtot * 32;
        scatter_cycles<<<(int)((thr + 255) / 256), 256, 0, stream>>>(cycle, csrc, cdst, S, Mtot);
    }

    const long long nWaves = E >> 4;
    const int blocks = (int)((nWaves + 7) / 8);
    gemm_bf16<1><<<blocks, 256, 0, stream>>>(x, Wf1, bond, eidx, E, S, nullptr, nullptr);
    gemm_bf16<2><<<blocks, 256, 0, stream>>>(S, Wf2, nullptr, nullptr, E, nullptr, out, stats);

    const long long total = E * DMODEL;
    const long long thr4  = total / 4;
    bn_gelu<<<(int)((thr4 + 255) / 256), 256, 0, stream>>>(out, stats, gamma, beta, total,
                                                           1.0f / (float)E);
}
